// IRLS_71622874628668
// MI455X (gfx1250) — compile-verified, matrix operand order checked by reading, not run on hardware
//
#include <hip/hip_runtime.h>

#define N_NODES 100000
#define N_EDGES 1600000
#define DIN 256
#define DHID 64
#define DOUT 40
#define PROP_STEP 16

typedef __attribute__((ext_vector_type(16))) _Float16 v16h;
typedef __attribute__((ext_vector_type(8)))  float    v8f;

// K index held by element j of a v16h A/B fragment for v_wmma_f32_16x16x32_f16,
// given the lane's half-select hsel = lane>>4 (CDNA5 ISA, 16-bit 16x32 layout):
//   lanes 0-15:  VGPR0..3 -> K 0..7,  VGPR4..7 -> K 16..23
//   lanes 16-31: VGPR0..3 -> K 8..15, VGPR4..7 -> K 24..31
__device__ __forceinline__ int kmap(int j, int hsel) {
    return (j < 8) ? (hsel * 8 + j) : (16 + hsel * 8 + (j - 8));
}

// -------- GEMM1: h = x @ W1 + b1   (one wave -> 16 rows x 64 cols) --------
__global__ void __launch_bounds__(32)
gnn_gemm1_wmma(const float* __restrict__ x, const float* __restrict__ W1,
               const float* __restrict__ b1, float* __restrict__ hout) {
    const int lane = threadIdx.x;
    const int l15  = lane & 15;
    const int hsel = lane >> 4;
    const int m0   = blockIdx.x * 16;

    v8f acc[4];
#pragma unroll
    for (int nt = 0; nt < 4; ++nt) acc[nt] = (v8f){};

    const float* xrow = x + (size_t)(m0 + l15) * DIN;

    for (int kt = 0; kt < DIN / 32; ++kt) {
        const int K0 = kt * 32;
        // A fragment: rows of x, converted f32 -> f16 (pairs are K-contiguous)
        v16h a;
#pragma unroll
        for (int t = 0; t < 8; ++t) {
            const int k = K0 + kmap(2 * t, hsel);
            float2 v = *(const float2*)(xrow + k);
            a[2 * t]     = (_Float16)v.x;
            a[2 * t + 1] = (_Float16)v.y;
        }
#pragma unroll
        for (int nt = 0; nt < 4; ++nt) {
            const int n = nt * 16 + l15;
            v16h b;
#pragma unroll
            for (int j = 0; j < 16; ++j) {
                b[j] = (_Float16)W1[(size_t)(K0 + kmap(j, hsel)) * DHID + n];
            }
            acc[nt] = __builtin_amdgcn_wmma_f32_16x16x32_f16(
                false, a, false, b, (short)0, acc[nt], false, false);
        }
    }
    // C layout: element r of v8f -> row M = r + 8*hsel, col N = l15
#pragma unroll
    for (int nt = 0; nt < 4; ++nt) {
        const int n = nt * 16 + l15;
        const float bias = b1[n];
#pragma unroll
        for (int r = 0; r < 8; ++r) {
            const int row = m0 + r + 8 * hsel;
            hout[(size_t)row * DHID + n] = acc[nt][r] + bias;
        }
    }
}

// -------- row_ptr build: lower_bound over sorted edge_row --------
__global__ void gnn_build_rowptr(const int* __restrict__ erow, int* __restrict__ rp) {
    const int n = blockIdx.x * blockDim.x + threadIdx.x;
    if (n > N_NODES) return;
    int lo = 0, hi = N_EDGES;
    while (lo < hi) {
        const int mid = (lo + hi) >> 1;
        if (erow[mid] < n) lo = mid + 1; else hi = mid;
    }
    rp[n] = lo;
}

// -------- propagation step: Ynew = 0.5*Yold + 0.5*A@Yold + 0.5*diag*h --------
// one 64-thread block per node; thread d owns feature d; edges staged in LDS
__global__ void __launch_bounds__(DHID)
gnn_prop_step(const float* __restrict__ Yold, const float* __restrict__ h,
              const float* __restrict__ diag, const int* __restrict__ rp,
              const int* __restrict__ ecol, const float* __restrict__ eval,
              float* __restrict__ Ynew) {
    const int n = blockIdx.x;
    const int d = threadIdx.x;
    const int s = rp[n];
    const int e = rp[n + 1];

    __shared__ int   scol[DHID];
    __shared__ float sval[DHID];

    float acc = 0.0f;
    for (int base = s; base < e; base += DHID) {
        const int i = base + d;
        if (i < e) { scol[d] = ecol[i]; sval[d] = eval[i]; }
        __syncthreads();
        const int cnt = min(DHID, e - base);
        for (int t = 0; t < cnt; ++t) {
            acc += sval[t] * Yold[(size_t)scol[t] * DHID + d];  // coalesced 256B gather
        }
        __syncthreads();
    }
    const size_t idx = (size_t)n * DHID + d;
    Ynew[idx] = 0.5f * Yold[idx] + 0.5f * acc + 0.5f * diag[n] * h[idx];
}

// -------- GEMM2: out = relu(Y) @ W2 + b2   (N padded 40 -> 48, masked) --------
__global__ void __launch_bounds__(32)
gnn_gemm2_wmma(const float* __restrict__ Y, const float* __restrict__ W2,
               const float* __restrict__ b2, float* __restrict__ out) {
    const int lane = threadIdx.x;
    const int l15  = lane & 15;
    const int hsel = lane >> 4;
    const int m0   = blockIdx.x * 16;

    v8f acc[3];
#pragma unroll
    for (int nt = 0; nt < 3; ++nt) acc[nt] = (v8f){};

    const float* yrow = Y + (size_t)(m0 + l15) * DHID;

    for (int kt = 0; kt < DHID / 32; ++kt) {
        const int K0 = kt * 32;
        v16h a;
#pragma unroll
        for (int t = 0; t < 8; ++t) {
            const int k = K0 + kmap(2 * t, hsel);
            float2 v = *(const float2*)(yrow + k);
            a[2 * t]     = (_Float16)fmaxf(v.x, 0.0f);   // fused ReLU
            a[2 * t + 1] = (_Float16)fmaxf(v.y, 0.0f);
        }
#pragma unroll
        for (int nt = 0; nt < 3; ++nt) {
            const int n = nt * 16 + l15;
            v16h b;
#pragma unroll
            for (int j = 0; j < 16; ++j) {
                const int kk = K0 + kmap(j, hsel);
                b[j] = (n < DOUT) ? (_Float16)W2[(size_t)kk * DOUT + n] : (_Float16)0.0f;
            }
            acc[nt] = __builtin_amdgcn_wmma_f32_16x16x32_f16(
                false, a, false, b, (short)0, acc[nt], false, false);
        }
    }
#pragma unroll
    for (int nt = 0; nt < 3; ++nt) {
        const int n = nt * 16 + l15;
        if (n < DOUT) {
            const float bias = b2[n];
#pragma unroll
            for (int r = 0; r < 8; ++r) {
                const int row = m0 + r + 8 * hsel;
                out[(size_t)row * DOUT + n] = acc[nt][r] + bias;
            }
        }
    }
}

extern "C" void kernel_launch(void* const* d_in, const int* in_sizes, int n_in,
                              void* d_out, int out_size, void* d_ws, size_t ws_size,
                              hipStream_t stream) {
    const float* x    = (const float*)d_in[0];
    const int*   erow = (const int*)  d_in[1];
    const int*   ecol = (const int*)  d_in[2];
    const float* eval = (const float*)d_in[3];
    const float* diag = (const float*)d_in[4];
    const float* W1   = (const float*)d_in[5];
    const float* b1   = (const float*)d_in[6];
    const float* W2   = (const float*)d_in[7];
    const float* b2   = (const float*)d_in[8];
    float* out = (float*)d_out;

    // workspace layout: h | Ya | Yb | row_ptr   (~77 MB, L2-resident working set)
    char*  ws  = (char*)d_ws;
    size_t off = 0;
    float* h  = (float*)(ws + off); off += (size_t)N_NODES * DHID * sizeof(float);
    float* Ya = (float*)(ws + off); off += (size_t)N_NODES * DHID * sizeof(float);
    float* Yb = (float*)(ws + off); off += (size_t)N_NODES * DHID * sizeof(float);
    int*   rp = (int*)  (ws + off); off += (size_t)(N_NODES + 1) * sizeof(int);

    gnn_build_rowptr<<<(N_NODES + 256) / 256, 256, 0, stream>>>(erow, rp);
    gnn_gemm1_wmma<<<N_NODES / 16, 32, 0, stream>>>(x, W1, b1, h);

    const float* cur = h;   // Y0 = h
    float* dst = Ya;
    for (int it = 0; it < PROP_STEP; ++it) {
        gnn_prop_step<<<N_NODES, DHID, 0, stream>>>(cur, h, diag, rp, ecol, eval, dst);
        cur = dst;
        dst = (dst == Ya) ? Yb : Ya;
    }

    gnn_gemm2_wmma<<<N_NODES / 16, 32, 0, stream>>>(cur, W2, b2, out);
}
